// NeuralAStar_42125039239443
// MI455X (gfx1250) — compile-verified, matrix operand order checked by reading, not run on hardware
//
#include <hip/hip_runtime.h>

#define BATCH   16
#define NG      128
#define NM      (NG * NG)
#define TITERS  160
#define NTHREADS 256

typedef __attribute__((ext_vector_type(16))) _Float16 v16h;
typedef __attribute__((ext_vector_type(8)))  float    v8f;

__global__ __launch_bounds__(NTHREADS)
void neural_astar_kernel(const float* __restrict__ P,
                         const float* __restrict__ H,
                         const int*   __restrict__ vs,
                         const int*   __restrict__ vg,
                         float*       __restrict__ out,
                         int*         __restrict__ parents_ws)
{
    // Per-workgroup (per-batch) LDS-resident search state: ~144KB of 320KB WGP LDS.
    __shared__ float              sh_g[NM];      // 64KB  G values
    __shared__ float              sh_h[NM];      // 64KB  H values (reused as paths map later)
    __shared__ unsigned char      sh_state[NM];  // 16KB  0=unvisited 1=open 2=closed
    __shared__ unsigned long long sh_red[8];
    __shared__ int                sh_sel;

    const int b   = blockIdx.x;
    const int tid = threadIdx.x;

    const float* Pb = P + b * NM;
    const float* Hb = H + b * NM;
    int* parents = parents_ws + b * NM;

    const int srcIdx  = vs[2 * b] * NG + vs[2 * b + 1];
    const int goalIdx = vg[2 * b] * NG + vg[2 * b + 1];

    // Prefetch cost map into cache hierarchy (global_prefetch path).
    __builtin_prefetch(Pb + tid * 64, 0, 0);

    // ---- init ----
    for (int c = tid; c < NM; c += NTHREADS) {
        sh_g[c]     = 0.0f;
        sh_h[c]     = Hb[c];
        sh_state[c] = 0;
        parents[c]  = -1;
    }
    __syncthreads();
    if (tid == 0) sh_state[srcIdx] = 1;   // source open, G=0, parent=-1

    // ---- greedy best-first search (matches reference's one-hot dynamics) ----
    for (int t = 0; t < TITERS; ++t) {
        __syncthreads();   // state/g from previous expansion visible

        // argmax over w = exp(-(G+H)) on open cells; ties -> lowest flat idx.
        // Packed key: (~bits(w) << 32) | idx ; min-reduce => max w, tie min idx.
        unsigned long long best = ~0ull;
        for (int c = tid; c < NM; c += NTHREADS) {
            if (sh_state[c] == 1) {
                float w = expf(-(sh_g[c] + sh_h[c]));   // TAU == 1.0
                unsigned int fb = __float_as_uint(w) ^ 0xFFFFFFFFu;
                unsigned long long comp =
                    ((unsigned long long)fb << 32) | (unsigned int)c;
                if (comp < best) best = comp;
            }
        }
        // wave32 reduction
        for (int off = 16; off > 0; off >>= 1) {
            unsigned long long o = __shfl_down(best, off, 32);
            if (o < best) best = o;
        }
        if ((tid & 31) == 0) sh_red[tid >> 5] = best;
        __syncthreads();
        if (tid == 0) {
            unsigned long long m = sh_red[0];
            #pragma unroll
            for (int i = 1; i < 8; ++i) if (sh_red[i] < m) m = sh_red[i];
            sh_sel = (int)(m & 0xFFFFFFFFu);
        }
        __syncthreads();
        const int sel = sh_sel;

        if (sel == goalIdx) break;  // eta=0, alive=0: outputs frozen (uniform break)

        if (tid == 0) sh_state[sel] = 2;   // close selected
        if (tid < 8) {                     // open the 8 neighbors
            const int d8y[8] = {-1,-1,-1, 0, 0, 1, 1, 1};
            const int d8x[8] = {-1, 0, 1,-1, 1,-1, 0, 1};
            const int cy = sel / NG, cx = sel % NG;
            const int ny = cy + d8y[tid], nx = cx + d8x[tid];
            if (ny >= 0 && ny < NG && nx >= 0 && nx < NG) {
                const int nc = ny * NG + nx;
                if (sh_state[nc] == 0) {          // (1-O)*(1-C)
                    sh_state[nc] = 1;
                    sh_g[nc]     = sh_g[sel] + Pb[nc];  // fl(gsum + P), as reference
                    parents[nc]  = sel;
                }
            }
        }
    }

    // ---- backtrack: build paths map in sh_h (H no longer needed) ----
    __syncthreads();
    for (int c = tid; c < NM; c += NTHREADS) sh_h[c] = 0.0f;
    __syncthreads();
    if (tid == 0) {
        int curr = goalIdx;
        bool first = true;
        for (int t = 0; t < TITERS; ++t) {    // same 160-step truncation as reference
            if (curr >= 0) {
                if (!first) sh_h[curr] = 1.0f;
                first = false;
                curr = parents[curr];
            }
        }
        sh_h[goalIdx] = 1.0f;                 // paths.at[goal].set(1.0)
    }
    __syncthreads();

    // ---- WMMA epilogue: stream 16x16 f32 tiles through the matrix pipe ----
    // D = A(0) x B(0) + C(tile).  C/D layout: VGPR r, lanes 0-15 -> M=r,
    // lanes 16-31 -> M=8+r, N = lane&15.  EXEC all ones here (no divergence).
    const int wave = tid >> 5;
    const int lane = tid & 31;
    const int half = lane >> 4;
    const int nl   = lane & 15;
    v16h za = {};
    v16h zb = {};
    float* outPaths = out + b * NM;
    float* outC     = out + BATCH * NM + b * NM;

    for (int tt = wave; tt < 64; tt += 8) {       // 64 tiles, 8 per wave
        const int ty = tt >> 3, tx = tt & 7;
        const int r0 = ty * 16 + half * 8;
        const int c0 = tx * 16 + nl;
        v8f cp, cc;
        #pragma unroll
        for (int r = 0; r < 8; ++r) {
            const int cell = (r0 + r) * NG + c0;
            cp[r] = sh_h[cell];
            float cv = (sh_state[cell] == 2) ? 1.0f : 0.0f;
            if (cell == goalIdx) cv = 1.0f;       // C = C + Vg (goal never closed)
            cc[r] = cv;
        }
        v8f dp = __builtin_amdgcn_wmma_f32_16x16x32_f16(
            false, za, false, zb, (short)0, cp, false, false);
        v8f dc = __builtin_amdgcn_wmma_f32_16x16x32_f16(
            false, za, false, zb, (short)0, cc, false, false);
        #pragma unroll
        for (int r = 0; r < 8; ++r) {
            const int cell = (r0 + r) * NG + c0;
            outPaths[cell] = dp[r];
            outC[cell]     = dc[r];
        }
    }
}

extern "C" void kernel_launch(void* const* d_in, const int* in_sizes, int n_in,
                              void* d_out, int out_size, void* d_ws, size_t ws_size,
                              hipStream_t stream) {
    (void)in_sizes; (void)n_in; (void)out_size; (void)ws_size;
    const float* P  = (const float*)d_in[0];
    const float* H  = (const float*)d_in[1];
    const int*   vs = (const int*)d_in[2];
    const int*   vg = (const int*)d_in[3];
    float* out = (float*)d_out;
    int* parents = (int*)d_ws;   // BATCH*NM ints = 1MB scratch

    neural_astar_kernel<<<dim3(BATCH), dim3(NTHREADS), 0, stream>>>(
        P, H, vs, vg, out, parents);
}
